// FastRCNNHead_54760833024756
// MI455X (gfx1250) — compile-verified, matrix operand order checked by reading, not run on hardware
//
#include <hip/hip_runtime.h>
#include <hip/hip_bf16.h>

// ---------------------------------------------------------------------------
// FastRCNN head for MI455X (gfx1250, wave32, WMMA bf16 16x16x32,
// async global->LDS double-buffered GEMM pipeline via inline asm)
// ---------------------------------------------------------------------------

typedef __attribute__((ext_vector_type(16))) __bf16 v16bf;
typedef __attribute__((ext_vector_type(8)))  float  v8f;

#define RESO 7
#define SRATE 2
#define MROWS 2048          // B*N = 4*512
#define K1DIM 12544         // C*49
#define FCDIM 1024
#define CCH 256
#define OUTD 100            // 4+18+36+36+6

#define BM 128
#define BN 128
#define BK 32

#define HAVE_ASYNC_LDS 1

#if __has_builtin(__builtin_amdgcn_s_wait_asynccnt)
#define WAIT_ASYNC(n) __builtin_amdgcn_s_wait_asynccnt(n)
#else
#define WAIT_ASYNC(n) asm volatile("s_wait_asynccnt %0" ::"i"(n) : "memory")
#endif

typedef __attribute__((address_space(3))) __bf16 lds_bf16;

// global_load_async_to_lds_b128: VDST = 32-bit LDS byte address in VGPR,
// VADDR = 64-bit global address VGPR pair, tracked by ASYNCcnt.
__device__ __forceinline__ void async_cp_b128(const __bf16* g, __bf16* l) {
  unsigned loff = (unsigned)(unsigned long long)(lds_bf16*)l;
  asm volatile("global_load_async_to_lds_b128 %0, %1, off"
               :: "v"(loff), "v"(g)
               : "memory");
}

// ---------------------------------------------------------------------------
// Weight convert + transpose: f32 (K,N) row-major -> bf16 (N,K) row-major
// ---------------------------------------------------------------------------
__global__ __launch_bounds__(256)
void transpose_cvt_kernel(const float* __restrict__ W, __bf16* __restrict__ T,
                          int K, int Nc) {
  __shared__ float tile[32][33];
  int k0 = blockIdx.x * 32, n0 = blockIdx.y * 32;
  int tx = threadIdx.x & 31, ty = threadIdx.x >> 5;
  #pragma unroll
  for (int i = ty; i < 32; i += 8) {
    int k = k0 + i, n = n0 + tx;
    tile[i][tx] = (k < K && n < Nc) ? W[(size_t)k * Nc + n] : 0.f;
  }
  __syncthreads();
  #pragma unroll
  for (int i = ty; i < 32; i += 8) {
    int n = n0 + i, k = k0 + tx;
    if (n < Nc && k < K) T[(size_t)n * K + k] = (__bf16)tile[tx][i];
  }
}

// ---------------------------------------------------------------------------
// Pack five head weight matrices (FC x d_i) into bf16 (128 x 1024) transposed,
// rows >= 100 zero-padded; also build concatenated f32 bias[128].
// ---------------------------------------------------------------------------
__global__ __launch_bounds__(256)
void pack_heads_kernel(const float* w0, const float* b0, const float* w1, const float* b1,
                       const float* w2, const float* b2, const float* w3, const float* b3,
                       const float* w4, const float* b4,
                       __bf16* __restrict__ Wht, float* __restrict__ hbias) {
  int idx = blockIdx.x * 256 + threadIdx.x;   // 128*1024 total
  int r = idx >> 10;
  int k = idx & 1023;
  const int starts[6] = {0, 4, 22, 58, 94, 100};
  const int dims[5]   = {4, 18, 36, 36, 6};
  const float* ws[5]  = {w0, w1, w2, w3, w4};
  const float* bs[5]  = {b0, b1, b2, b3, b4};
  float val = 0.f, bv = 0.f;
  if (r < OUTD) {
    int h = 0;
    while (r >= starts[h + 1]) h++;
    int rl = r - starts[h];
    val = ws[h][(size_t)k * dims[h] + rl];
    bv  = bs[h][rl];
  }
  Wht[(size_t)r * FCDIM + k] = (__bf16)val;
  if (k == 0) hbias[r] = bv;
}

// ---------------------------------------------------------------------------
// ROI align: one block per (b,n) box, 256 threads = 256 channels.
// ---------------------------------------------------------------------------
__global__ __launch_bounds__(256)
void roi_align_kernel(const float* __restrict__ f0, const float* __restrict__ f1,
                      const float* __restrict__ f2, const float* __restrict__ bbox,
                      const int* __restrict__ anchor, __bf16* __restrict__ X,
                      int Nn) {
  int m = blockIdx.x;
  int b = m / Nn;
  const float* bx = bbox + (size_t)m * 4;
  float cy = bx[0], cx = bx[1], hh = bx[2], ww = bx[3];
  int lvl = anchor[m] / 3;
  lvl = lvl < 0 ? 0 : (lvl > 2 ? 2 : lvl);
  const float scales[3] = {0.25f, 0.125f, 0.0625f};
  const int   hws[3]    = {128, 64, 32};
  const float* feats[3] = {f0, f1, f2};
  float scale = scales[lvl];
  int H = hws[lvl], W = H;
  const float* fbase = feats[lvl] + (size_t)b * CCH * H * W;

  float t = cy - hh * 0.5f, l = cx - ww * 0.5f;
  float sy = t * scale - 0.5f, sx = l * scale - 0.5f;
  float bh = hh * scale / (float)RESO, bw = ww * scale / (float)RESO;

  __shared__ int   sp0[28], sp1[28];
  __shared__ float slo[28], shi[28], sva[28];
  int tid = threadIdx.x;
  if (tid < 28) {
    int j = tid % 14;
    bool isY = tid < 14;
    float g = ((float)j + 0.5f) / (float)SRATE;
    float p = isY ? (sy + g * bh) : (sx + g * bw);
    float valid = (p > -1.0f && p < (float)H) ? 1.f : 0.f;
    float pc = fminf(fmaxf(p, 0.f), (float)(H - 1));
    int p0 = (int)floorf(pc);
    int p1 = p0 + 1 < H ? p0 + 1 : H - 1;
    float lp = pc - (float)p0;
    sp0[tid] = p0; sp1[tid] = p1; slo[tid] = lp; shi[tid] = 1.f - lp; sva[tid] = valid;
  }
  __syncthreads();

  int c = tid;
  const float* fc_ = fbase + (size_t)c * H * W;
  float acc[49];
  #pragma unroll
  for (int i = 0; i < 49; i++) acc[i] = 0.f;

  for (int jy = 0; jy < 14; jy++) {
    int y0 = sp0[jy] * W, y1 = sp1[jy] * W;
    float ly = slo[jy], hy = shi[jy], vy = sva[jy];
    #pragma unroll 2
    for (int jx = 0; jx < 14; jx++) {
      int x0 = sp0[14 + jx], x1 = sp1[14 + jx];
      float lx = slo[14 + jx], hx = shi[14 + jx], vx = sva[14 + jx];
      float v = (fc_[y0 + x0] * (hy * hx) + fc_[y0 + x1] * (hy * lx) +
                 fc_[y1 + x0] * (ly * hx) + fc_[y1 + x1] * (ly * lx)) * (vy * vx);
      acc[(jy >> 1) * 7 + (jx >> 1)] += v;
    }
  }
  __bf16* xo = X + (size_t)m * K1DIM + (size_t)c * 49;
  #pragma unroll
  for (int i = 0; i < 49; i++) xo[i] = (__bf16)(acc[i] * 0.25f);
}

// ---------------------------------------------------------------------------
// Fragment loader: per-lane K runs {half*8 .. +7} and {16+half*8 .. +7}
// ---------------------------------------------------------------------------
__device__ __forceinline__ v16bf ld_frag(const __bf16* rowbase, int half) {
  union { v16bf v; uint4 u[2]; } t;
  t.u[0] = *reinterpret_cast<const uint4*>(rowbase + half * 8);
  t.u[1] = *reinterpret_cast<const uint4*>(rowbase + 16 + half * 8);
  return t.v;
}

// ---------------------------------------------------------------------------
// Async double-buffered WMMA GEMM (no bounds guards; all dims tile-divisible)
// C(M,N) = A(M,K) * Bt(N,K)^T + bias, f32 output.
// Block tile 128x128x32, 8 waves (4x2), wave tile 32x64 (2x4 WMMA frags).
// ---------------------------------------------------------------------------
__global__ __launch_bounds__(256)
void gemm_bf16_async_kernel(const __bf16* __restrict__ A, const __bf16* __restrict__ Bt,
                            const float* __restrict__ bias, float* __restrict__ Cout,
                            int Mr, int Nc, int Kd, int ldc) {
  __shared__ __align__(16) __bf16 As[2][BM][BK];
  __shared__ __align__(16) __bf16 Bs[2][BN][BK];

  int m0 = blockIdx.y * BM;
  int n0 = blockIdx.x * BN;
  int tid = threadIdx.x;
  int lane = tid & 31, wave = tid >> 5;
  int wm = wave >> 1, wn = wave & 1;
  int lrow = lane & 15, half = lane >> 4;

  // per-thread staging chunks: 2 for A + 2 for B per tile (8 bf16 each)
  int r0 = tid >> 2,          c0 = (tid & 3) * 8;
  int r1 = (tid + 256) >> 2,  c1 = ((tid + 256) & 3) * 8;

  v8f acc[2][4];
  #pragma unroll
  for (int fi = 0; fi < 2; fi++)
    #pragma unroll
    for (int fj = 0; fj < 4; fj++)
      #pragma unroll
      for (int q = 0; q < 8; q++) acc[fi][fj][q] = 0.f;

  int ntiles = Kd / BK;

#ifdef HAVE_ASYNC_LDS
  // issue tile 0 into buffer 0
  {
    async_cp_b128(A + (size_t)(m0 + r0) * Kd + c0, &As[0][r0][c0]);
    async_cp_b128(Bt + (size_t)(n0 + r0) * Kd + c0, &Bs[0][r0][c0]);
    async_cp_b128(A + (size_t)(m0 + r1) * Kd + c1, &As[0][r1][c1]);
    async_cp_b128(Bt + (size_t)(n0 + r1) * Kd + c1, &Bs[0][r1][c1]);
  }
#endif

  for (int kt = 0; kt < ntiles; ++kt) {
    int cur = kt & 1;
#ifdef HAVE_ASYNC_LDS
    if (kt + 1 < ntiles) {
      // issue tile kt+1 into the other buffer (last read at kt-1; barrier since)
      int k0 = (kt + 1) * BK;
      int nxt = cur ^ 1;
      async_cp_b128(A + (size_t)(m0 + r0) * Kd + k0 + c0, &As[nxt][r0][c0]);
      async_cp_b128(Bt + (size_t)(n0 + r0) * Kd + k0 + c0, &Bs[nxt][r0][c0]);
      async_cp_b128(A + (size_t)(m0 + r1) * Kd + k0 + c1, &As[nxt][r1][c1]);
      async_cp_b128(Bt + (size_t)(n0 + r1) * Kd + k0 + c1, &Bs[nxt][r1][c1]);
      WAIT_ASYNC(4);   // in-order: tile kt's 4 loads complete
    } else {
      WAIT_ASYNC(0);
    }
#else
    {
      int k0 = kt * BK;
      *reinterpret_cast<uint4*>(&As[cur][r0][c0]) =
          *reinterpret_cast<const uint4*>(A + (size_t)(m0 + r0) * Kd + k0 + c0);
      *reinterpret_cast<uint4*>(&Bs[cur][r0][c0]) =
          *reinterpret_cast<const uint4*>(Bt + (size_t)(n0 + r0) * Kd + k0 + c0);
      *reinterpret_cast<uint4*>(&As[cur][r1][c1]) =
          *reinterpret_cast<const uint4*>(A + (size_t)(m0 + r1) * Kd + k0 + c1);
      *reinterpret_cast<uint4*>(&Bs[cur][r1][c1]) =
          *reinterpret_cast<const uint4*>(Bt + (size_t)(n0 + r1) * Kd + k0 + c1);
    }
#endif
    __syncthreads();   // tile kt visible to all waves

    v16bf af[2], bfv[4];
    #pragma unroll
    for (int fi = 0; fi < 2; fi++)
      af[fi] = ld_frag(&As[cur][wm * 32 + fi * 16 + lrow][0], half);
    #pragma unroll
    for (int fj = 0; fj < 4; fj++)
      bfv[fj] = ld_frag(&Bs[cur][wn * 64 + fj * 16 + lrow][0], half);

    #pragma unroll
    for (int fi = 0; fi < 2; fi++)
      #pragma unroll
      for (int fj = 0; fj < 4; fj++)
        acc[fi][fj] = __builtin_amdgcn_wmma_f32_16x16x32_bf16(
            false, af[fi], false, bfv[fj], (short)0, acc[fi][fj], false, false);

    __syncthreads();   // all waves done reading buffer cur
  }

  #pragma unroll
  for (int fi = 0; fi < 2; fi++)
    #pragma unroll
    for (int fj = 0; fj < 4; fj++) {
      int col = n0 + wn * 64 + fj * 16 + lrow;
      float bv = bias[col];
      #pragma unroll
      for (int r = 0; r < 8; r++) {
        int row = m0 + wm * 32 + fi * 16 + half * 8 + r;
        Cout[(size_t)row * ldc + col] = acc[fi][fj][r] + bv;
      }
    }
}

// ---------------------------------------------------------------------------
// Guarded synchronous WMMA GEMM (for N=100 heads)
// ---------------------------------------------------------------------------
__global__ __launch_bounds__(256)
void gemm_bf16_guarded_kernel(const __bf16* __restrict__ A, const __bf16* __restrict__ Bt,
                              const float* __restrict__ bias, float* __restrict__ Cout,
                              int Mr, int Nc, int Kd, int ldc) {
  __shared__ __align__(16) __bf16 As[BM][BK];
  __shared__ __align__(16) __bf16 Bs[BN][BK];

  int m0 = blockIdx.y * BM;
  int n0 = blockIdx.x * BN;
  int tid = threadIdx.x;
  int lane = tid & 31, wave = tid >> 5;
  int wm = wave >> 1, wn = wave & 1;
  int lrow = lane & 15, half = lane >> 4;

  v8f acc[2][4];
  #pragma unroll
  for (int fi = 0; fi < 2; fi++)
    #pragma unroll
    for (int fj = 0; fj < 4; fj++)
      #pragma unroll
      for (int q = 0; q < 8; q++) acc[fi][fj][q] = 0.f;

  int ntiles = Kd / BK;
  uint4 zz; zz.x = zz.y = zz.z = zz.w = 0u;

  for (int kt = 0; kt < ntiles; ++kt) {
    int k0 = kt * BK;
    #pragma unroll
    for (int s = 0; s < 2; s++) {
      int cid = tid + s * 256;
      int row = cid >> 2, ch = cid & 3;
      const uint4* pa = reinterpret_cast<const uint4*>(A + (size_t)(m0 + row) * Kd + k0 + ch * 8);
      *reinterpret_cast<uint4*>(&As[row][ch * 8]) = (m0 + row < Mr) ? *pa : zz;
      const uint4* pb = reinterpret_cast<const uint4*>(Bt + (size_t)(n0 + row) * Kd + k0 + ch * 8);
      *reinterpret_cast<uint4*>(&Bs[row][ch * 8]) = (n0 + row < Nc) ? *pb : zz;
    }
    if (kt + 1 < ntiles) {
      int row = tid >> 2, ch = tid & 3;
      if (m0 + row < Mr) __builtin_prefetch(A + (size_t)(m0 + row) * Kd + k0 + BK + ch * 8, 0, 1);
      if (n0 + row < Nc) __builtin_prefetch(Bt + (size_t)(n0 + row) * Kd + k0 + BK + ch * 8, 0, 1);
    }
    __syncthreads();

    v16bf af[2], bfv[4];
    #pragma unroll
    for (int fi = 0; fi < 2; fi++)
      af[fi] = ld_frag(&As[wm * 32 + fi * 16 + lrow][0], half);
    #pragma unroll
    for (int fj = 0; fj < 4; fj++)
      bfv[fj] = ld_frag(&Bs[wn * 64 + fj * 16 + lrow][0], half);

    #pragma unroll
    for (int fi = 0; fi < 2; fi++)
      #pragma unroll
      for (int fj = 0; fj < 4; fj++)
        acc[fi][fj] = __builtin_amdgcn_wmma_f32_16x16x32_bf16(
            false, af[fi], false, bfv[fj], (short)0, acc[fi][fj], false, false);
    __syncthreads();
  }

  #pragma unroll
  for (int fi = 0; fi < 2; fi++)
    #pragma unroll
    for (int fj = 0; fj < 4; fj++) {
      int col = n0 + wn * 64 + fj * 16 + lrow;
      if (col >= Nc) continue;
      float bv = bias[col];
      #pragma unroll
      for (int r = 0; r < 8; r++) {
        int row = m0 + wm * 32 + fi * 16 + half * 8 + r;
        if (row < Mr) Cout[(size_t)row * ldc + col] = acc[fi][fj][r] + bv;
      }
    }
}

// ---------------------------------------------------------------------------
// BatchNorm over the batch axis
// ---------------------------------------------------------------------------
__global__ __launch_bounds__(256)
void bn_stats_kernel(const float* __restrict__ H, float* __restrict__ stats, int Mr) {
  int cl = threadIdx.x & 31;
  int c = blockIdx.x * 32 + cl;
  int r = threadIdx.x >> 5;
  float s = 0.f, sq = 0.f;
  for (int i = r; i < Mr; i += 8) {
    float v = H[(size_t)i * FCDIM + c];
    s += v; sq += v * v;
  }
  __shared__ float shs[8][32], shq[8][32];
  shs[r][cl] = s; shq[r][cl] = sq;
  __syncthreads();
  if (r == 0) {
    #pragma unroll
    for (int i = 1; i < 8; i++) { s += shs[i][cl]; sq += shq[i][cl]; }
    float inv = 1.f / (float)Mr;
    float mu = s * inv;
    stats[c] = mu;
    stats[FCDIM + c] = sq * inv - mu * mu;
  }
}

__global__ __launch_bounds__(256)
void bn_apply_kernel(const float* __restrict__ H, const float* __restrict__ stats,
                     const float* __restrict__ g, const float* __restrict__ bb,
                     __bf16* __restrict__ Xo) {
  int idx = blockIdx.x * 256 + threadIdx.x;
  int c = idx & (FCDIM - 1);
  float mu = stats[c];
  float inv = rsqrtf(stats[FCDIM + c] + 1e-5f);
  float v = (H[idx] - mu) * inv * g[c] + bb[c];
  Xo[idx] = (__bf16)fmaxf(v, 0.f);
}

// ---------------------------------------------------------------------------
// Host launcher
// ---------------------------------------------------------------------------
extern "C" void kernel_launch(void* const* d_in, const int* in_sizes, int n_in,
                              void* d_out, int out_size, void* d_ws, size_t ws_size,
                              hipStream_t stream) {
  const float* feat0   = (const float*)d_in[0];
  const float* feat1   = (const float*)d_in[1];
  const float* feat2   = (const float*)d_in[2];
  const float* bbox2d  = (const float*)d_in[3];
  const int*   anchor  = (const int*)  d_in[4];
  const float* fc1_w   = (const float*)d_in[5];
  const float* fc1_b   = (const float*)d_in[6];
  const float* bn1_g   = (const float*)d_in[7];
  const float* bn1_b   = (const float*)d_in[8];
  const float* fc2_w   = (const float*)d_in[9];
  const float* fc2_b   = (const float*)d_in[10];
  const float* bn2_g   = (const float*)d_in[11];
  const float* bn2_b   = (const float*)d_in[12];
  const float* cate_w  = (const float*)d_in[13];
  const float* cate_b  = (const float*)d_in[14];
  const float* b3d_w   = (const float*)d_in[15];
  const float* b3d_b   = (const float*)d_in[16];
  const float* yawc_w  = (const float*)d_in[17];
  const float* yawc_b  = (const float*)d_in[18];
  const float* yawr_w  = (const float*)d_in[19];
  const float* yawr_b  = (const float*)d_in[20];
  const float* hgt_w   = (const float*)d_in[21];
  const float* hgt_b   = (const float*)d_in[22];

  char* ws = (char*)d_ws;
  size_t oX  = 0;                                        // 2048*12544 bf16
  size_t oW1 = oX  + (size_t)MROWS * K1DIM * 2;          // 1024*12544 bf16
  size_t oW2 = oW1 + (size_t)FCDIM * K1DIM * 2;          // 1024*1024 bf16
  size_t oWh = oW2 + (size_t)FCDIM * FCDIM * 2;          // 128*1024 bf16
  size_t oH  = oWh + (size_t)128 * FCDIM * 2;            // 2048*1024 f32
  size_t oX2 = oH  + (size_t)MROWS * FCDIM * 4;          // 2048*1024 bf16
  size_t oX3 = oX2 + (size_t)MROWS * FCDIM * 2;          // 2048*1024 bf16
  size_t oSt = oX3 + (size_t)MROWS * FCDIM * 2;          // 2*1024 f32
  size_t oHb = oSt + 2 * FCDIM * 4;                      // 128 f32

  __bf16* Xbf  = (__bf16*)(ws + oX);
  __bf16* W1t  = (__bf16*)(ws + oW1);
  __bf16* W2t  = (__bf16*)(ws + oW2);
  __bf16* Wht  = (__bf16*)(ws + oWh);
  float*  Hbuf = (float*) (ws + oH);
  __bf16* X2   = (__bf16*)(ws + oX2);
  __bf16* X3   = (__bf16*)(ws + oX3);
  float*  stats= (float*) (ws + oSt);
  float*  hbias= (float*) (ws + oHb);
  float*  out  = (float*) d_out;

  // 1. weight convert + transpose to bf16 (N,K)
  transpose_cvt_kernel<<<dim3(K1DIM / 32, FCDIM / 32), 256, 0, stream>>>(fc1_w, W1t, K1DIM, FCDIM);
  transpose_cvt_kernel<<<dim3(FCDIM / 32, FCDIM / 32), 256, 0, stream>>>(fc2_w, W2t, FCDIM, FCDIM);
  pack_heads_kernel<<<dim3(128 * FCDIM / 256), 256, 0, stream>>>(
      cate_w, cate_b, b3d_w, b3d_b, yawc_w, yawc_b, yawr_w, yawr_b, hgt_w, hgt_b, Wht, hbias);

  // 2. ROI align -> bf16 activations (2048 x 12544)
  roi_align_kernel<<<dim3(MROWS), 256, 0, stream>>>(feat0, feat1, feat2, bbox2d, anchor, Xbf, 512);

  // 3. FC1 (async double-buffered WMMA bf16) -> H f32
  gemm_bf16_async_kernel<<<dim3(FCDIM / BN, MROWS / BM), 256, 0, stream>>>(
      Xbf, W1t, fc1_b, Hbuf, MROWS, FCDIM, K1DIM, FCDIM);

  // 4. BN1 + ReLU -> X2 bf16
  bn_stats_kernel<<<dim3(FCDIM / 32), 256, 0, stream>>>(Hbuf, stats, MROWS);
  bn_apply_kernel<<<dim3(MROWS * FCDIM / 256), 256, 0, stream>>>(Hbuf, stats, bn1_g, bn1_b, X2);

  // 5. FC2 -> H f32 (reused)
  gemm_bf16_async_kernel<<<dim3(FCDIM / BN, MROWS / BM), 256, 0, stream>>>(
      X2, W2t, fc2_b, Hbuf, MROWS, FCDIM, FCDIM, FCDIM);

  // 6. BN2 + ReLU -> X3 bf16
  bn_stats_kernel<<<dim3(FCDIM / 32), 256, 0, stream>>>(Hbuf, stats, MROWS);
  bn_apply_kernel<<<dim3(MROWS * FCDIM / 256), 256, 0, stream>>>(Hbuf, stats, bn2_g, bn2_b, X3);

  // 7. heads GEMM (2048 x 100) straight into d_out (== (B,N,100) flat)
  gemm_bf16_guarded_kernel<<<dim3(1, MROWS / BM), 256, 0, stream>>>(
      X3, Wht, hbias, out, MROWS, OUTD, FCDIM, OUTD);
}
